// MonaOpClipAdapter_67843303408339
// MI455X (gfx1250) — matrix operand order repaired and checked by reading, not compiled, not run on hardware
//
#include <hip/hip_runtime.h>
#include <hip/hip_bf16.h>
#include <math.h>

// ---------------------------------------------------------------------------
// MonaOpClipAdapter for MI455X (gfx1250, wave32, WMMA bf16 16x16x32)
// - fragments: pairs of b128 loads (LDS or global), ISA-exact layouts
// - branch-free GELU/sigmoid (native exp + rcp)
// - LDS tile-transpose epilogues for coalesced NCHW / row-major traffic
// - LN stats fused into down-proj (x read once); row-norm fused into up-proj;
//   gate GEMM + cosine-attention GEMM merged (shared A-tiles)
// ---------------------------------------------------------------------------

typedef __bf16 bf16_t;
typedef __attribute__((ext_vector_type(16))) __bf16 v16bf;
typedef __attribute__((ext_vector_type(8)))  __bf16 v8bf;
typedef __attribute__((ext_vector_type(8)))  float  v8f;

#define WMMA_BF16(a, b, c) \
  __builtin_amdgcn_wmma_f32_16x16x32_bf16(false, (a), false, (b), (short)0, (c), false, false)

constexpr int Bn   = 8;
constexpr int Cn   = 768;
constexpr int HWp  = 4096;          // 64*64
constexpr int NPIX = Bn * HWp;      // 32768
constexpr int BOT  = 64;
constexpr int PD   = 96;
constexpr int KP   = 32;

// branch-free exact-GELU: A&S 7.1.26 erf approx (|err|<1.5e-7), native exp
__device__ __forceinline__ float gelu_f(float x) {
  float ax = fabsf(x) * 0.70710678118654752f;
  float t  = __builtin_amdgcn_rcpf(1.0f + 0.3275911f * ax);
  float p  = t * (0.254829592f + t * (-0.284496736f + t * (1.421413741f +
             t * (-1.453152027f + t * 1.061405429f))));
  float er = 1.0f - p * __expf(-ax * ax);
  er = copysignf(er, x);
  return 0.5f * x * (1.0f + er);
}
__device__ __forceinline__ float sigm_f(float x) {
  return __builtin_amdgcn_rcpf(1.0f + __expf(-x));
}
__device__ __forceinline__ v8f vzero8() {
  v8f z = {0.f, 0.f, 0.f, 0.f, 0.f, 0.f, 0.f, 0.f};
  return z;
}
// Build a 16-element bf16 fragment from two 16-byte chunks (2 x b128 loads).
struct bf16x16 { uint4 lo, hi; };
__device__ __forceinline__ v16bf frag2(const void* p0, const void* p1) {
  bf16x16 t;
  t.lo = *(const uint4*)p0;
  t.hi = *(const uint4*)p1;
  return __builtin_bit_cast(v16bf, t);
}
// A-fragment (16x32, MxK): halves 0-7 at k0+(lane>=16?8:0), halves 8-15 at +16
__device__ __forceinline__ v16bf a_frag_row(const bf16_t* row_k0, int lane) {
  const bf16_t* p = row_k0 + ((lane >> 4) << 3);
  return frag2(p, p + 16);
}
// B-fragment (32x16, KxN) from n-major row: halves 0-7 at k0+(lane>=16?8:0), halves 8-15 at +16
__device__ __forceinline__ v16bf b_frag_row(const bf16_t* row_k0, int lane) {
  const bf16_t* p = row_k0 + ((lane >> 4) << 3);
  return frag2(p, p + 16);
}
__device__ __forceinline__ unsigned pack_bf2(float a, float b) {
  union { __bf16 h[2]; unsigned u; } t;
  t.h[0] = (__bf16)a;
  t.h[1] = (__bf16)b;
  return t.u;
}
__device__ __forceinline__ float samp2d(const float* __restrict__ src, int yi, int xi) {
  bool ok = (yi >= 0) & (yi < 64) & (xi >= 0) & (xi < 64);
  int yc = yi < 0 ? 0 : (yi > 63 ? 63 : yi);
  int xc = xi < 0 ? 0 : (xi > 63 ? 63 : xi);
  return ok ? src[yc * 64 + xc] : 0.0f;
}

// ---------------------------------------------------------------------------
// Weight prep: bf16 conversions / transposes (n-major so B-fragments are
// k-contiguous b128 loads).
// ---------------------------------------------------------------------------
__global__ __launch_bounds__(256) void k_dcw(const float* __restrict__ w,
                                             bf16_t* __restrict__ out) {
  // dc_w (o,c,3,3) flat == (o, c*9+t): already n-major over K=576
  int i = blockIdx.x * 256 + threadIdx.x;
  if (i >= 64 * 576) return;
  out[i] = (bf16_t)w[i];
}
__global__ __launch_bounds__(256) void k_prep_wu(const float* __restrict__ Wu,
                                                 bf16_t* __restrict__ WuT) {
  int i = blockIdx.x * 256 + threadIdx.x;   // WuT[n*64+k] = Wu[k*768+n]
  if (i >= 768 * 64) return;
  int n = i >> 6, k = i & 63;
  WuT[i] = (bf16_t)Wu[k * 768 + n];
}
__global__ __launch_bounds__(256) void k_prep_wg1(const float* __restrict__ Wg1,
                                                  bf16_t* __restrict__ Wg1T) {
  int i = blockIdx.x * 256 + threadIdx.x;   // Wg1T[n*768+k] = Wg1[k*192+n]
  if (i >= 192 * 768) return;
  int n = i / 768, k = i - n * 768;
  Wg1T[i] = (bf16_t)Wg1[k * 192 + n];
}

// ---------------------------------------------------------------------------
// K2: fused LN-stats + LayerNorm + down-proj GEMM + bias + GELU
//     block = 128 thr (4 waves) = 32 pixels x 64 outputs
//     x is streamed ONCE: raw bf16 slab parked in LDS (48KB) while fp32
//     stats accumulate; LN applied in-place; K-loop runs off the slab.
// ---------------------------------------------------------------------------
__global__ __launch_bounds__(128) void k_down(const float* __restrict__ x,
                                              const float* __restrict__ lng,
                                              const float* __restrict__ lnb,
                                              const float* __restrict__ Wd,
                                              const float* __restrict__ bd,
                                              float* __restrict__ xc) {
  __shared__ __align__(16) bf16_t Axl[32][768];   // 48KB A-slab
  __shared__ __align__(16) bf16_t BsT[64][32];    // [n][k] per K-step
  __shared__ float sg[768], sb[768];
  __shared__ float psum[128], pssq[128];
  __shared__ float smean[32], srstd[32];
  int tid = threadIdx.x, lane = tid & 31, wave = tid >> 5;
  int pbase = blockIdx.x * 32;
  int b = pbase >> 12, hwbase = pbase & 4095;

  for (int i = tid; i < 768; i += 128) { sg[i] = lng[i]; sb[i] = lnb[i]; }

  // phase A: stream x once (coalesced over 32 pixels), park bf16, fp32 stats
  float s = 0.f, ss = 0.f;
  for (int i = tid; i < 32 * 384; i += 128) {
    int pl = i & 31, kp = i >> 5;       // pl == tid&31 (fixed per thread)
    int c0 = 2 * kp;
    float v0 = x[((size_t)(b * Cn + c0)) * HWp + (hwbase + pl)];
    float v1 = x[((size_t)(b * Cn + c0 + 1)) * HWp + (hwbase + pl)];
    *(unsigned*)&Axl[pl][c0] = pack_bf2(v0, v1);
    s += v0 + v1;
    ss += v0 * v0 + v1 * v1;
  }
  psum[tid] = s;
  pssq[tid] = ss;
  __syncthreads();
  if (tid < 32) {
    float sm = psum[tid] + psum[tid + 32] + psum[tid + 64] + psum[tid + 96];
    float sq = pssq[tid] + pssq[tid + 32] + pssq[tid + 64] + pssq[tid + 96];
    float m = sm * (1.0f / Cn);
    float var = sq * (1.0f / Cn) - m * m;
    smean[tid] = m;
    srstd[tid] = rsqrtf(var + 1e-5f);
  }
  __syncthreads();
  // phase B: apply LayerNorm in place (each thread rewrites its own words)
  for (int i = tid; i < 32 * 384; i += 128) {
    int pl = i & 31, kp = i >> 5;
    int c0 = 2 * kp;
    float m = smean[pl], r = srstd[pl];
    union { unsigned u; __bf16 h[2]; } tt;
    tt.u = *(unsigned*)&Axl[pl][c0];
    float v0 = ((float)tt.h[0] - m) * r * sg[c0] + sb[c0];
    float v1 = ((float)tt.h[1] - m) * r * sg[c0 + 1] + sb[c0 + 1];
    *(unsigned*)&Axl[pl][c0] = pack_bf2(v0, v1);
  }

  // wave -> tile mapping: pg = 16-pixel group, nh = 32-column half
  int pg = wave >> 1, nh = wave & 1;
  v8f acc[2];
  acc[0] = vzero8();
  acc[1] = vzero8();
  for (int k0 = 0; k0 < Cn; k0 += 32) {
    __syncthreads();
    for (int i = tid; i < 1024; i += 128) {
      int n = i & 63, kp2 = i >> 6;
      float w0 = Wd[(k0 + 2 * kp2) * BOT + n];
      float w1 = Wd[(k0 + 2 * kp2 + 1) * BOT + n];
      *(unsigned*)&BsT[n][2 * kp2] = pack_bf2(w0, w1);
    }
    __syncthreads();
    v16bf a = a_frag_row(&Axl[pg * 16 + (lane & 15)][k0], lane);
#pragma unroll
    for (int t = 0; t < 2; ++t) {
      v16bf bb = b_frag_row(&BsT[nh * 32 + t * 16 + (lane & 15)][0], lane);
      acc[t] = WMMA_BF16(a, bb, acc[t]);
    }
  }
#pragma unroll
  for (int t = 0; t < 2; ++t) {
    int n = nh * 32 + t * 16 + (lane & 15);
#pragma unroll
    for (int v = 0; v < 8; ++v) {
      int m = ((lane >> 4) << 3) + v;
      int pl = pg * 16 + m;
      float val = gelu_f(acc[t][v] + bd[n]);
      xc[((size_t)(b * BOT + n)) * HWp + (hwbase + pl)] = val;
    }
  }
}

// ---------------------------------------------------------------------------
// K3: 3x3 conv for offsets: in = concat(xc, irf) 128ch -> 18ch
// ---------------------------------------------------------------------------
__global__ __launch_bounds__(256) void k_offconv(const float* __restrict__ xc,
                                                 const float* __restrict__ irf,
                                                 const float* __restrict__ w,
                                                 const float* __restrict__ bias,
                                                 float* __restrict__ offs) {
  int idx = blockIdx.x * 256 + threadIdx.x;   // (b*18+o)*4096 + hw
  int hw = idx & 4095;
  int bo = idx >> 12;
  int b = bo / 18, o = bo - b * 18;
  int y = hw >> 6, xx = hw & 63;
  float acc = bias[o];
  for (int ci = 0; ci < 128; ++ci) {
    const float* src = (ci < 64) ? (xc + ((size_t)(b * 64 + ci)) * HWp)
                                 : (irf + ((size_t)(b * 64 + (ci - 64))) * HWp);
    const float* wr = w + ((size_t)(o * 128 + ci)) * 9;
#pragma unroll
    for (int ky = 0; ky < 3; ++ky) {
      int yy = y + ky - 1;
      if (yy < 0 || yy > 63) continue;
#pragma unroll
      for (int kx = 0; kx < 3; ++kx) {
        int xxx = xx + kx - 1;
        if (xxx < 0 || xxx > 63) continue;
        acc += src[yy * 64 + xxx] * wr[ky * 3 + kx];
      }
    }
  }
  offs[idx] = acc;
}

// ---------------------------------------------------------------------------
// K4: modulation MLP on irf: 64 -> gelu(16) -> sigmoid(64), per pixel
// ---------------------------------------------------------------------------
__global__ __launch_bounds__(128) void k_mod(const float* __restrict__ irf,
                                             const float* __restrict__ w1,
                                             const float* __restrict__ b1,
                                             const float* __restrict__ w2,
                                             const float* __restrict__ b2,
                                             float* __restrict__ mmod) {
  int p = blockIdx.x * 128 + threadIdx.x;
  int b = p >> 12, hw = p & 4095;
  float v[64];
#pragma unroll
  for (int c = 0; c < 64; ++c) v[c] = irf[((size_t)(b * 64 + c)) * HWp + hw];
  float h[16];
#pragma unroll
  for (int o = 0; o < 16; ++o) {
    float s = b1[o];
#pragma unroll
    for (int c = 0; c < 64; ++c) s += v[c] * w1[o * 64 + c];
    h[o] = gelu_f(s);
  }
  for (int o = 0; o < 64; ++o) {
    float s = b2[o];
#pragma unroll
    for (int j = 0; j < 16; ++j) s += h[j] * w2[o * 16 + j];
    mmod[((size_t)(b * 64 + o)) * HWp + hw] = sigm_f(s);
  }
}

// ---------------------------------------------------------------------------
// K5: deformable conv as im2col (bilinear gather -> LDS) + WMMA GEMM
//     block = 128 thr (4 waves) = 16 pixels x 64 outputs, K = 576
//     fused: x_fused = x_def * (1 + alpha*m)
// ---------------------------------------------------------------------------
__global__ __launch_bounds__(128) void k_deform(const float* __restrict__ xc,
                                                const float* __restrict__ offs,
                                                const bf16_t* __restrict__ wkb,
                                                const float* __restrict__ mmod,
                                                const float* __restrict__ alpha,
                                                float* __restrict__ xfused) {
  __shared__ float sy0[144], sx0[144], swy[144], swx[144];
  __shared__ __align__(16) bf16_t As[16][576];
  int tid = threadIdx.x, lane = tid & 31, wave = tid >> 5;
  int pbase = blockIdx.x * 16;
  int b = pbase >> 12, hwbase = pbase & 4095;

  for (int i = tid; i < 144; i += 128) {
    int pl = i / 9, t = i - pl * 9;
    int hw = hwbase + pl;
    int y = hw >> 6, xx = hw & 63;
    float dy = offs[((size_t)(b * 18 + 2 * t)) * HWp + hw];
    float dx = offs[((size_t)(b * 18 + 2 * t + 1)) * HWp + hw];
    float py = (float)(y - 1 + t / 3) + dy;
    float px = (float)(xx - 1 + t % 3) + dx;
    float y0 = floorf(py), x0 = floorf(px);
    sy0[i] = y0; sx0[i] = x0; swy[i] = py - y0; swx[i] = px - x0;
  }
  __syncthreads();

  // im2col: two K-elements per loop iteration, packed 32-bit LDS store
  for (int i = tid; i < 16 * 288; i += 128) {
    int pl = i / 288, kp = i - pl * 288;
    float vv[2];
#pragma unroll
    for (int u = 0; u < 2; ++u) {
      int kk = 2 * kp + u;
      int c = kk / 9, t = kk - c * 9;
      int e9 = pl * 9 + t;
      int y0 = (int)sy0[e9], x0 = (int)sx0[e9];
      float wy = swy[e9], wx = swx[e9];
      const float* src = xc + ((size_t)(b * 64 + c)) * HWp;
      float g00 = samp2d(src, y0, x0);
      float g01 = samp2d(src, y0, x0 + 1);
      float g10 = samp2d(src, y0 + 1, x0);
      float g11 = samp2d(src, y0 + 1, x0 + 1);
      vv[u] = g00 * (1 - wy) * (1 - wx) + g01 * (1 - wy) * wx +
              g10 * wy * (1 - wx) + g11 * wy * wx;
    }
    *(unsigned*)&As[pl][2 * kp] = pack_bf2(vv[0], vv[1]);
  }
  __syncthreads();

  int o = wave * 16 + (lane & 15);
  const bf16_t* brow = wkb + (size_t)o * 576;   // n-major weights
  v8f acc = vzero8();
  for (int k0 = 0; k0 < 576; k0 += 32) {
    v16bf a = a_frag_row(&As[lane & 15][k0], lane);
    v16bf bb = b_frag_row(brow + k0, lane);
    acc = WMMA_BF16(a, bb, acc);
  }
  float al = alpha[0];
#pragma unroll
  for (int v = 0; v < 8; ++v) {
    int m = ((lane >> 4) << 3) + v;
    int hw = hwbase + m;
    float mv = mmod[((size_t)(b * 64 + o)) * HWp + hw];
    xfused[((size_t)(b * 64 + o)) * HWp + hw] = acc[v] * (1.0f + al * mv);
  }
}

// ---------------------------------------------------------------------------
// K6: up-proj GEMM (32768x64 @ 64x768) + residual: x_res1 = xp + scale*x_up
//     + fused per-pixel sumsq reduction (row norms) into ssq[]
//     grid.y = 3 column groups of 256; block = 128 (4 waves x 64 cols)
// ---------------------------------------------------------------------------
__global__ __launch_bounds__(128) void k_up(const float* __restrict__ x,
                                            const float* __restrict__ xfused,
                                            const bf16_t* __restrict__ WuT,
                                            const float* __restrict__ bu,
                                            const float* __restrict__ scale,
                                            bf16_t* __restrict__ xres,
                                            float* __restrict__ ssq) {
  __shared__ __align__(16) bf16_t As[16][64];
  __shared__ float tbuf[4][16][20];   // per-wave transpose tile (pad 20: 16B rows)
  __shared__ float sred[16];
  int tid = threadIdx.x, lane = tid & 31, wave = tid >> 5;
  int pbase = blockIdx.x * 16;
  int b = pbase >> 12, hwbase = pbase & 4095;
  if (tid < 16) sred[tid] = 0.f;
  for (int i = tid; i < 512; i += 128) {
    int pl = i & 15, kp = i >> 4;   // kp: 0..31
    int c0 = 2 * kp;
    float v0 = xfused[((size_t)(b * 64 + c0)) * HWp + hwbase + pl];
    float v1 = xfused[((size_t)(b * 64 + c0 + 1)) * HWp + hwbase + pl];
    *(unsigned*)&As[pl][c0] = pack_bf2(v0, v1);
  }
  __syncthreads();
  int nbase = blockIdx.y * 256 + wave * 64;
  v8f acc[4];
#pragma unroll
  for (int t = 0; t < 4; ++t) acc[t] = vzero8();
#pragma unroll
  for (int k0 = 0; k0 < 64; k0 += 32) {
    v16bf a = a_frag_row(&As[lane & 15][k0], lane);
#pragma unroll
    for (int t = 0; t < 4; ++t) {
      int n = nbase + t * 16 + (lane & 15);
      v16bf bb = b_frag_row(WuT + (size_t)n * 64 + k0, lane);
      acc[t] = WMMA_BF16(a, bb, acc[t]);
    }
  }
  float sc = scale[0];
  float part[8];
#pragma unroll
  for (int v = 0; v < 8; ++v) part[v] = 0.f;
  for (int t = 0; t < 4; ++t) {
    int n = nbase + t * 16 + (lane & 15);
    // phase 1: residual add in WMMA layout, park in per-wave LDS tile
#pragma unroll
    for (int v = 0; v < 8; ++v) {
      int m = ((lane >> 4) << 3) + v;
      int hw = hwbase + m;
      float xv = x[((size_t)(b * Cn + n)) * HWp + hw];
      float val = xv + sc * (acc[t][v] + bu[n]);
      part[v] += val * val;
      tbuf[wave][m][lane & 15] = val;
    }
    // phase 2: coalesced bf16 row store (16B per lane; DS in-order per wave)
    {
      int pl = lane >> 1, ch = (lane & 1) * 8;
      const float* row = &tbuf[wave][pl][ch];
      uint4 qq;
      qq.x = pack_bf2(row[0], row[1]);
      qq.y = pack_bf2(row[2], row[3]);
      qq.z = pack_bf2(row[4], row[5]);
      qq.w = pack_bf2(row[6], row[7]);
      *(uint4*)(xres + (size_t)(pbase + pl) * Cn + nbase + t * 16 + ch) = qq;
    }
  }
  // per-pixel sumsq: local -> LDS -> one global atomic per pixel
#pragma unroll
  for (int v = 0; v < 8; ++v) {
    int m = ((lane >> 4) << 3) + v;
    atomicAdd(&sred[m], part[v]);
  }
  __syncthreads();
  if (tid < 16) atomicAdd(&ssq[pbase + tid], sred[tid]);
}

// ---------------------------------------------------------------------------
// K8a: prompt keys/values pk = tf@Wk + bk, pv = tf@Wv + bv (tiny GEMMs)
// ---------------------------------------------------------------------------
__global__ __launch_bounds__(128) void k_pkv(const float* __restrict__ tf,
                                             const float* __restrict__ Wk,
                                             const float* __restrict__ bk,
                                             const float* __restrict__ Wv,
                                             const float* __restrict__ bv,
                                             float* __restrict__ pk,
                                             float* __restrict__ pv) {
  int i = blockIdx.x * 128 + threadIdx.x;
  if (i >= KP * Cn) return;
  int k = i / Cn, c = i - k * Cn;
  float sk = bk[c], sv = bv[c];
  for (int j = 0; j < PD; ++j) {
    float t = tf[k * PD + j];
    sk += t * Wk[j * Cn + c];
    sv += t * Wv[j * Cn + c];
  }
  pk[i] = sk;
  pv[i] = sv;
}

// ---------------------------------------------------------------------------
// K8b: normalize pk rows -> bf16 pkn (k-major); pv -> bf16 transposed (c-major)
// ---------------------------------------------------------------------------
__global__ __launch_bounds__(128) void k_pknorm(const float* __restrict__ pk,
                                                const float* __restrict__ pv,
                                                bf16_t* __restrict__ pknb,
                                                bf16_t* __restrict__ pvT) {
  __shared__ float red[128];
  int k = blockIdx.x, tid = threadIdx.x;
  float s = 0.f;
  for (int c = tid; c < Cn; c += 128) {
    float v = pk[k * Cn + c];
    s += v * v;
  }
  red[tid] = s;
  __syncthreads();
  for (int st = 64; st > 0; st >>= 1) {
    if (tid < st) red[tid] += red[tid + st];
    __syncthreads();
  }
  float inv = __builtin_amdgcn_rcpf(fmaxf(sqrtf(red[0]), 1e-12f));
  for (int c = tid; c < Cn; c += 128) {
    pknb[k * Cn + c] = (bf16_t)(pk[k * Cn + c] * inv);
    pvT[c * KP + k]  = (bf16_t)pv[k * Cn + c];
  }
}

// ---------------------------------------------------------------------------
// K9+K10 merged: gate GEMM (192 cols, waves 0-2) + cosine GEMM (32 keys,
// wave 3) share the same x_res1 A-tiles (staged once per K-step).
// Epilogues: gate -> sigmoid(gelu(h)@Wg2+bg2); attn -> softmax -> bf16
// ---------------------------------------------------------------------------
__global__ __launch_bounds__(128) void k_attn_gate(const bf16_t* __restrict__ xres,
                                                   const float* __restrict__ ssq,
                                                   const bf16_t* __restrict__ pknb,
                                                   const float* __restrict__ temperature,
                                                   const bf16_t* __restrict__ Wg1T,
                                                   const float* __restrict__ bg1,
                                                   const float* __restrict__ Wg2,
                                                   const float* __restrict__ bg2,
                                                   float* __restrict__ gout,
                                                   bf16_t* __restrict__ attn) {
  __shared__ __align__(16) bf16_t As[16][32];
  __shared__ float gred[16];
  __shared__ float cs[16][32];
  int tid = threadIdx.x, lane = tid & 31, wave = tid >> 5;
  int pbase = blockIdx.x * 16;
  if (tid < 16) gred[tid] = 0.f;
  bool isAttn = (wave == 3);
  const bf16_t* brow0 = pknb + (size_t)(lane & 15) * Cn;
  const bf16_t* brow1 = pknb + (size_t)((lane & 15) + 16) * Cn;
  v8f acc[4];
#pragma unroll
  for (int t = 0; t < 4; ++t) acc[t] = vzero8();
  for (int k0 = 0; k0 < Cn; k0 += 32) {
    __syncthreads();
    // 16 rows x 32 k = 64 uint4 chunks, contiguous in global
    for (int i = tid; i < 64; i += 128) {
      int pl = i & 15, qc = i >> 4;
      *(uint4*)&As[pl][qc * 8] =
          *(const uint4*)(xres + (size_t)(pbase + pl) * Cn + k0 + qc * 8);
    }
    __syncthreads();
    v16bf a = a_frag_row(&As[lane & 15][0], lane);
    if (!isAttn) {
#pragma unroll
      for (int t = 0; t < 4; ++t) {
        int n = wave * 64 + t * 16 + (lane & 15);
        v16bf bb = b_frag_row(Wg1T + (size_t)n * Cn + k0, lane);
        acc[t] = WMMA_BF16(a, bb, acc[t]);
      }
    } else {
      v16bf b0 = b_frag_row(brow0 + k0, lane);
      v16bf b1 = b_frag_row(brow1 + k0, lane);
      acc[0] = WMMA_BF16(a, b0, acc[0]);
      acc[1] = WMMA_BF16(a, b1, acc[1]);
    }
  }
  if (!isAttn) {
#pragma unroll
    for (int t = 0; t < 4; ++t) {
      int n = wave * 64 + t * 16 + (lane & 15);
      float w2 = Wg2[n];
#pragma unroll
      for (int v = 0; v < 8; ++v) {
        int m = ((lane >> 4) << 3) + v;
        float h = gelu_f(acc[t][v] + bg1[n]);
        atomicAdd(&gred[m], h * w2);
      }
    }
  } else {
    float tinv = __builtin_amdgcn_rcpf(temperature[0]);
#pragma unroll
    for (int v = 0; v < 8; ++v) {
      int m = ((lane >> 4) << 3) + v;
      float inv = __builtin_amdgcn_rcpf(fmaxf(sqrtf(ssq[pbase + m]), 1e-12f));
      float sc = inv * tinv;
      cs[m][lane & 15] = acc[0][v] * sc;
      cs[m][16 + (lane & 15)] = acc[1][v] * sc;
    }
  }
  __syncthreads();
  if (tid < 16) gout[pbase + tid] = sigm_f(gred[tid] + bg2[0]);
  if (isAttn && lane < 16) {
    int row = lane;
    float mx = -1e30f;
    for (int k = 0; k < 32; ++k) mx = fmaxf(mx, cs[row][k]);
    float ev[32], sum = 0.f;
    for (int k = 0; k < 32; ++k) {
      ev[k] = __expf(cs[row][k] - mx);
      sum += ev[k];
    }
    float is = __builtin_amdgcn_rcpf(sum);
    for (int k = 0; k < 32; ++k)
      attn[(size_t)(pbase + row) * KP + k] = (bf16_t)(ev[k] * is);
  }
}

// ---------------------------------------------------------------------------
// K11: fused x_prompted = attn@pv (single K=32 WMMA per 16x16 tile) +
//      final out = x + scale*(x_res1 + g*x_prompted)
//      LDS tile-transpose -> hw-contiguous coalesced x read / out write
// ---------------------------------------------------------------------------
__global__ __launch_bounds__(128) void k_final(const float* __restrict__ x,
                                               const bf16_t* __restrict__ xres,
                                               const bf16_t* __restrict__ attn,
                                               const bf16_t* __restrict__ pvT,
                                               const float* __restrict__ gout,
                                               const float* __restrict__ scale,
                                               float* __restrict__ out) {
  __shared__ float tbuf[4][16][20];   // per-wave transpose tile
  int tid = threadIdx.x, lane = tid & 31, wave = tid >> 5;
  int pbase = blockIdx.x * 16;
  int b = pbase >> 12, hwbase = pbase & 4095;
  v16bf a = a_frag_row(attn + (size_t)(pbase + (lane & 15)) * KP, lane);
  float sc = scale[0];
  float gv[8];
#pragma unroll
  for (int v = 0; v < 8; ++v) {
    int m = ((lane >> 4) << 3) + v;
    gv[v] = gout[pbase + m];
  }
  for (int t = 0; t < 12; ++t) {
    int n = wave * 192 + t * 16 + (lane & 15);
    v16bf bb = b_frag_row(pvT + (size_t)n * KP, lane);
    v8f acc = WMMA_BF16(a, bb, vzero8());
    // phase 1: x_res1 + g*prompt in WMMA layout, park in per-wave LDS tile
#pragma unroll
    for (int v = 0; v < 8; ++v) {
      int m = ((lane >> 4) << 3) + v;
      int p = pbase + m;
      float xr = (float)xres[(size_t)p * Cn + n];
      tbuf[wave][m][lane & 15] = xr + gv[v] * acc[v];
    }
    // phase 2: coalesced NCHW read-modify-write (lanes span hw; DS in-order)
    {
      int m = lane & 15;
      int chalf = lane >> 4;           // 0 or 1
#pragma unroll
      for (int u = 0; u < 8; ++u) {
        int cl = 2 * u + chalf;
        float val = tbuf[wave][m][cl];
        size_t gi = ((size_t)(b * Cn + wave * 192 + t * 16 + cl)) * HWp + hwbase + m;
        out[gi] = x[gi] + sc * val;
      }
    }
  }
}

// ---------------------------------------------------------------------------
// host launcher
// ---------------------------------------------------------------------------
extern "C" void kernel_launch(void* const* d_in, const int* in_sizes, int n_in,
                              void* d_out, int out_size, void* d_ws, size_t ws_size,
                              hipStream_t stream) {
  const float* x     = (const float*)d_in[0];
  const float* irf   = (const float*)d_in[1];
  const float* tf    = (const float*)d_in[2];
  const float* lng   = (const float*)d_in[3];
  const float* lnb   = (const float*)d_in[4];
  const float* Wd    = (const float*)d_in[5];
  const float* bd    = (const float*)d_in[6];
  const float* Wu    = (const float*)d_in[7];
  const float* bu    = (const float*)d_in[8];
  const float* off_w = (const float*)d_in[9];
  const float* off_b = (const float*)d_in[10];
  const float* mw1   = (const float*)d_in[11];
  const float* mb1   = (const float*)d_in[12];
  const float* mw2   = (const float*)d_in[13];
  const float* mb2   = (const float*)d_in[14];
  const float* dcw   = (const float*)d_in[15];
  const float* alpha = (const float*)d_in[16];
  const float* Wk    = (const float*)d_in[17];
  const float* bk    = (const float*)d_in[18];
  const float* Wv    = (const float*)d_in[19];
  const float* bv    = (const float*)d_in[20];
  const float* temp  = (const float*)d_in[21];
  const float* Wg1   = (const float*)d_in[22];
  const float* bg1   = (const float*)d_in[23];
  const float* Wg2   = (const float*)d_in[24];
  const float* bg2   = (const float*)d_in[25];
  const float* scale = (const float*)d_in[26];
  float* out = (float*)d_out;

  char* ws = (char*)d_ws;
  size_t off = 0;
  auto alloc = [&](size_t bytes) -> void* {
    off = (off + 255) & ~(size_t)255;
    void* p = ws + off;
    off += bytes;
    return p;
  };
  float*  xc     = (float*)alloc(sizeof(float) * (size_t)Bn * BOT * HWp);
  float*  offs   = (float*)alloc(sizeof(float) * (size_t)Bn * 18 * HWp);
  float*  mmod   = (float*)alloc(sizeof(float) * (size_t)Bn * BOT * HWp);
  float*  xfused = (float*)alloc(sizeof(float) * (size_t)Bn * BOT * HWp);
  bf16_t* dcwb   = (bf16_t*)alloc(sizeof(bf16_t) * 64 * 576);
  bf16_t* WuT    = (bf16_t*)alloc(sizeof(bf16_t) * 768 * 64);
  bf16_t* Wg1T   = (bf16_t*)alloc(sizeof(bf16_t) * 192 * 768);
  bf16_t* xres   = (bf16_t*)alloc(sizeof(bf16_t) * (size_t)NPIX * Cn);
  float*  ssq    = (float*)alloc(sizeof(float) * NPIX);
  float*  pk     = (float*)alloc(sizeof(float) * KP * Cn);
  float*  pv     = (float*)alloc(sizeof(float) * KP * Cn);
  bf16_t* pknb   = (bf16_t*)alloc(sizeof(bf16_t) * KP * Cn);
  bf16_t* pvT    = (bf16_t*)alloc(sizeof(bf16_t) * Cn * KP);
  bf16_t* attn   = (bf16_t*)alloc(sizeof(bf16_t) * (size_t)NPIX * KP);
  float*  gout   = (float*)alloc(sizeof(float) * NPIX);

  k_dcw<<<(64 * 576 + 255) / 256, 256, 0, stream>>>(dcw, dcwb);
  k_prep_wu<<<(768 * 64 + 255) / 256, 256, 0, stream>>>(Wu, WuT);
  k_prep_wg1<<<(192 * 768 + 255) / 256, 256, 0, stream>>>(Wg1, Wg1T);
  k_down<<<NPIX / 32, 128, 0, stream>>>(x, lng, lnb, Wd, bd, xc);
  k_offconv<<<NPIX * 18 / 256, 256, 0, stream>>>(xc, irf, off_w, off_b, offs);
  k_mod<<<NPIX / 128, 128, 0, stream>>>(irf, mw1, mb1, mw2, mb2, mmod);
  k_deform<<<NPIX / 16, 128, 0, stream>>>(xc, offs, dcwb, mmod, alpha, xfused);
  hipMemsetAsync(ssq, 0, sizeof(float) * NPIX, stream);
  dim3 gup(NPIX / 16, 3);
  k_up<<<gup, 128, 0, stream>>>(x, xfused, WuT, bu, scale, xres, ssq);
  k_pkv<<<(KP * Cn + 127) / 128, 128, 0, stream>>>(tf, Wk, bk, Wv, bv, pk, pv);
  k_pknorm<<<KP, 128, 0, stream>>>(pk, pv, pknb, pvT);
  k_attn_gate<<<NPIX / 16, 128, 0, stream>>>(xres, ssq, pknb, temp, Wg1T, bg1,
                                             Wg2, bg2, gout, attn);
  k_final<<<NPIX / 16, 128, 0, stream>>>(x, xres, attn, pvT, gout, scale, out);

  (void)in_sizes; (void)n_in; (void)out_size; (void)ws_size;
}
